// Attention_20968030339090
// MI455X (gfx1250) — matrix operand order repaired and checked by reading, not compiled, not run on hardware
//
#include <hip/hip_runtime.h>
#include <hip/hip_bf16.h>
#include <stdint.h>

typedef __attribute__((ext_vector_type(16))) __bf16 v16bf;
typedef __attribute__((ext_vector_type(8)))  float  v8f;

// ---------------- bf16 helpers ----------------
__device__ __forceinline__ __bf16 f32_bf16(float f) {
    union { float f; uint32_t u; } x; x.f = f;
    uint32_t r = x.u + 0x7FFFu + ((x.u >> 16) & 1u);   // round-to-nearest-even
    unsigned short h = (unsigned short)(r >> 16);
    return __builtin_bit_cast(__bf16, h);
}

__device__ __forceinline__ v8f wmma_bf16(v16bf a, v16bf b, v8f c) {
    // D(f32 16x16) = A(16x32 bf16) * B(32x16 bf16) + C
    return __builtin_amdgcn_wmma_f32_16x16x32_bf16(
        false, a, false, b, (short)0, c, false, false);
}

// ---------------- fragment loaders (wave32; K = 16*(lane/16) + element, same order in A and B) ----
// A: 16(M) x 32(K) bf16, row-major source, leading dim ld (elements).
// lane: M = lane&15 ; VGPR pair p holds K = (lane/16)*16 + p*2 (+1)
__device__ __forceinline__ v16bf frag_a_load(const __bf16* src, int ld) {
    const int lane = threadIdx.x & 31;
    const int m = lane & 15, hi = lane >> 4;
    v16bf a;
#pragma unroll
    for (int p = 0; p < 8; ++p) {
        const int k = (hi << 4) + (p << 1);
        a[2 * p]     = src[m * ld + k];
        a[2 * p + 1] = src[m * ld + k + 1];
    }
    return a;
}

// B: 32(K) x 16(N) bf16 from (k,n) row-major source: elem(k,n) = src[k*ld + n]
// lane: N = lane&15 ; lanes 0-15 hold K=0-15, lanes 16-31 hold K=16-31 (2 per VGPR)
__device__ __forceinline__ v16bf frag_b_kn(const __bf16* src, int ld) {
    const int lane = threadIdx.x & 31;
    const int n = lane & 15, hi = lane >> 4;
    v16bf b;
#pragma unroll
    for (int p = 0; p < 8; ++p) {
        const int k = (hi << 4) + (p << 1);
        b[2 * p]     = src[k * ld + n];
        b[2 * p + 1] = src[(k + 1) * ld + n];
    }
    return b;
}

// B from transposed (n,k) source: elem(k,n) = src[n*ld + k]   (e.g. K^T from key-major K)
__device__ __forceinline__ v16bf frag_b_nk(const __bf16* src, int ld) {
    const int lane = threadIdx.x & 31;
    const int n = lane & 15, hi = lane >> 4;
    v16bf b;
#pragma unroll
    for (int p = 0; p < 8; ++p) {
        const int k = (hi << 4) + (p << 1);
        b[2 * p]     = src[n * ld + k];
        b[2 * p + 1] = src[n * ld + k + 1];
    }
    return b;
}

// ---------------- constants ----------------
#define Bsz   8
#define Hdim  32
#define Wdim  32
#define Cdim  768
#define NH    12
#define HD    64
#define Ntok  1024          // H*W
#define BH    96            // B*NH

// ============================================================================
// Kernel 1: QKV GEMM  (8192 x 2304) = x(8192x768) @ qkv_w^T, fused transpose
//           to (3, B*nh, N, hd), q scaled by hd^-0.5, stored bf16.
// ============================================================================
__global__ __launch_bounds__(128) void qkv_kernel(
    const float* __restrict__ x, const float* __restrict__ qkv_w,
    const float* __restrict__ qkv_b,
    __bf16* __restrict__ qb, __bf16* __restrict__ kb, __bf16* __restrict__ vb)
{
    __shared__ __align__(16) __bf16 As[64][32];
    __shared__ __align__(16) __bf16 Bs[32][32];
    const int tid = threadIdx.x, wave = tid >> 5, lane = tid & 31;
    const int hi = lane >> 4, col = lane & 15;
    const int m0 = blockIdx.x * 64;
    const int n0 = blockIdx.y * 32;
    v8f acc0 = {}, acc1 = {};

    for (int k0 = 0; k0 < Cdim; k0 += 32) {
        // stage A (64x32) f32 -> bf16
#pragma unroll
        for (int i = 0; i < 4; ++i) {
            int j = tid * 4 + i;                // 512 float4 vectors
            int r = j >> 3, c = (j & 7) << 2;
            const float4 f = *(const float4*)&x[(size_t)(m0 + r) * Cdim + k0 + c];
            As[r][c]     = f32_bf16(f.x);
            As[r][c + 1] = f32_bf16(f.y);
            As[r][c + 2] = f32_bf16(f.z);
            As[r][c + 3] = f32_bf16(f.w);
        }
        // stage B (k,n) from W[n][k] f32 -> bf16
#pragma unroll
        for (int i = 0; i < 2; ++i) {
            int j = tid * 2 + i;                // 256 float4 vectors
            int n = j >> 3, kq = (j & 7) << 2;
            const float4 f = *(const float4*)&qkv_w[(size_t)(n0 + n) * Cdim + k0 + kq];
            Bs[kq][n]     = f32_bf16(f.x);
            Bs[kq + 1][n] = f32_bf16(f.y);
            Bs[kq + 2][n] = f32_bf16(f.z);
            Bs[kq + 3][n] = f32_bf16(f.w);
        }
        __syncthreads();
        v16bf a  = frag_a_load(&As[wave * 16][0], 32);
        v16bf b0 = frag_b_kn(&Bs[0][0], 32);
        v16bf b1 = frag_b_kn(&Bs[0][16], 32);
        acc0 = wmma_bf16(a, b0, acc0);
        acc1 = wmma_bf16(a, b1, acc1);
        __syncthreads();
    }

    // epilogue: split into q/k/v with transpose, q pre-scaled (before rel einsums)
#pragma unroll
    for (int s = 0; s < 2; ++s) {
        const v8f acc = s ? acc1 : acc0;
        const int n = n0 + s * 16 + col;        // 0..2303
        const float bias = qkv_b[n];
        const int t = n / Cdim;                 // 0=q 1=k 2=v
        const int rem = n % Cdim;
        const int h = rem >> 6, d = rem & 63;
        __bf16* dst = (t == 0) ? qb : ((t == 1) ? kb : vb);
        const float sc = (t == 0) ? 0.125f : 1.0f;   // hd^-0.5
#pragma unroll
        for (int r = 0; r < 8; ++r) {
            const int m = m0 + wave * 16 + r + (hi << 3);
            const int bidx = m >> 10, token = m & 1023;
            const float v = (acc[r] + bias) * sc;
            dst[((size_t)(bidx * NH + h) * Ntok + token) * HD + d] = f32_bf16(v);
        }
    }
}

// ============================================================================
// Kernel 2: rel_h / rel_w tables via WMMA.
//   mode 0 (rel_h): fixed h; rows=w;  out[bh,h,w,kh] = q . rel_pos_h[h-kh+31]
//   mode 1 (rel_w): fixed w; rows=h;  out[bh,h,w,kw] = q . rel_pos_w[w-kw+31]
// One wave per block, 32x32 output, K=64.
// ============================================================================
__global__ __launch_bounds__(32) void rel_kernel(
    const __bf16* __restrict__ qb, const float* __restrict__ rel_pos_h,
    const float* __restrict__ rel_pos_w,
    float* __restrict__ rel_h, float* __restrict__ rel_w)
{
    const int id = blockIdx.x;
    const int mode = id / (BH * 32);
    const int rem = id % (BH * 32);
    const int bh = rem >> 5;
    const int fx = rem & 31;
    const float* table = mode ? rel_pos_w : rel_pos_h;
    const int lane = threadIdx.x & 31;
    const int hi = lane >> 4, col = lane & 15;

    const __bf16* qbase = qb + (size_t)bh * Ntok * HD;
    const int ldA = mode ? (32 * HD) : HD;            // row stride between A rows
    const __bf16* abase = mode ? (qbase + fx * HD) : (qbase + (size_t)fx * 32 * HD);

    v8f acc[2][2] = {};
    for (int k0 = 0; k0 < HD; k0 += 32) {
        v16bf a0 = frag_a_load(abase + k0, ldA);
        v16bf a1 = frag_a_load(abase + (size_t)16 * ldA + k0, ldA);
#pragma unroll
        for (int ns = 0; ns < 2; ++ns) {
            const int n = ns * 16 + col;
            const float* trow = &table[(size_t)(fx - n + 31) * HD + k0];
            v16bf b;
#pragma unroll
            for (int p = 0; p < 8; ++p) {
                const int k = (hi << 4) + (p << 1);
                b[2 * p]     = f32_bf16(trow[k]);
                b[2 * p + 1] = f32_bf16(trow[k + 1]);
            }
            acc[0][ns] = wmma_bf16(a0, b, acc[0][ns]);
            acc[1][ns] = wmma_bf16(a1, b, acc[1][ns]);
        }
    }

    float* out = mode ? rel_w : rel_h;
#pragma unroll
    for (int ms = 0; ms < 2; ++ms)
#pragma unroll
        for (int ns = 0; ns < 2; ++ns)
#pragma unroll
            for (int r = 0; r < 8; ++r) {
                const int row = ms * 16 + r + (hi << 3);
                const int n = ns * 16 + col;
                const size_t idx = mode
                    ? ((((size_t)bh * 32 + row) * 32 + fx) * 32 + n)
                    : ((((size_t)bh * 32 + fx) * 32 + row) * 32 + n);
                out[idx] = acc[ms][ns][r];
            }
}

// ============================================================================
// Kernel 3: flash attention with decomposed rel-pos bias.
// Block = 4 waves, 64 queries; loop over 32 key-chunks of 32 (kh = chunk).
// Never materializes the 402MB attn tensor.
// ============================================================================
__global__ __launch_bounds__(128) void attn_kernel(
    const __bf16* __restrict__ qb, const __bf16* __restrict__ kb,
    const __bf16* __restrict__ vb,
    const float* __restrict__ relh, const float* __restrict__ relw,
    __bf16* __restrict__ ao)
{
    __shared__ __align__(16) __bf16 Ks[32][64];
    __shared__ __align__(16) __bf16 Vs[32][64];
    __shared__ __align__(16) __bf16 Ps[4][16][32];
    const int tid = threadIdx.x, wave = tid >> 5, lane = tid & 31;
    const int hi = lane >> 4, col = lane & 15;
    const int bh = blockIdx.x >> 4;
    const int qt = blockIdx.x & 15;
    const int m0 = qt * 64 + wave * 16;          // first query row of this wave

    const __bf16* qrow = qb + ((size_t)bh * Ntok + m0) * HD;
    const v16bf qa0 = frag_a_load(qrow, HD);
    const v16bf qa1 = frag_a_load(qrow + 32, HD);

    v8f o[4] = {};
    float mrow[8], lrow[8];
#pragma unroll
    for (int r = 0; r < 8; ++r) { mrow[r] = -1e30f; lrow[r] = 0.f; }

    for (int c = 0; c < 32; ++c) {               // key chunk == kh row
        // ---- stage K,V chunk (32x64 bf16 each) ----
        const size_t koff = ((size_t)bh * Ntok + c * 32) * HD;
#pragma unroll
        for (int i = 0; i < 2; ++i) {
            const int j = tid * 2 + i;           // 256 uint4 per matrix
            const int key = j >> 3, dv = (j & 7) << 3;
            *(uint4*)&Ks[key][dv] = *(const uint4*)&kb[koff + key * HD + dv];
            *(uint4*)&Vs[key][dv] = *(const uint4*)&vb[koff + key * HD + dv];
        }
        __syncthreads();

        // ---- S = Q @ K^T  (16 x 32) ----
        v8f s0 = {}, s1 = {};
        {
            v16bf b00 = frag_b_nk(&Ks[0][0],  HD);
            v16bf b01 = frag_b_nk(&Ks[0][32], HD);
            v16bf b10 = frag_b_nk(&Ks[16][0], HD);
            v16bf b11 = frag_b_nk(&Ks[16][32], HD);
            s0 = wmma_bf16(qa0, b00, s0); s0 = wmma_bf16(qa1, b01, s0);
            s1 = wmma_bf16(qa0, b10, s1); s1 = wmma_bf16(qa1, b11, s1);
        }

        // ---- add rel-pos bias + online softmax ----
#pragma unroll
        for (int r = 0; r < 8; ++r) {
            const int row = r + (hi << 3);
            const int token = m0 + row;
            const size_t rb = ((size_t)bh * Ntok + token) * 32;   // (bh,h,w,*)
            const float bh_ = relh[rb + c];                        // kh == c
            s0[r] += bh_ + relw[rb + col];                         // kw == col
            s1[r] += bh_ + relw[rb + 16 + col];

            float rm = fmaxf(s0[r], s1[r]);
#pragma unroll
            for (int ms = 1; ms <= 8; ms <<= 1) rm = fmaxf(rm, __shfl_xor(rm, ms, 32));
            const float mn = fmaxf(mrow[r], rm);
            const float alpha = __expf(mrow[r] - mn);
            mrow[r] = mn;
            const float p0 = __expf(s0[r] - mn);
            const float p1 = __expf(s1[r] - mn);
            float rs = p0 + p1;
#pragma unroll
            for (int ms = 1; ms <= 8; ms <<= 1) rs += __shfl_xor(rs, ms, 32);
            lrow[r] = lrow[r] * alpha + rs;
#pragma unroll
            for (int f = 0; f < 4; ++f) o[f][r] *= alpha;

            Ps[wave][row][col]      = f32_bf16(p0);   // transpose P via LDS
            Ps[wave][row][16 + col] = f32_bf16(p1);
        }
        asm volatile("s_wait_dscnt 0" ::: "memory");   // intra-wave LDS RAW

        // ---- O += P @ V ----
        const v16bf pa = frag_a_load(&Ps[wave][0][0], 32);
#pragma unroll
        for (int f = 0; f < 4; ++f) {
            v16bf vbf = frag_b_kn(&Vs[0][f * 16], HD);
            o[f] = wmma_bf16(pa, vbf, o[f]);
        }
        __syncthreads();
    }

    // ---- epilogue: O/l, transpose heads back to (B, token, C), store bf16 ----
    const int b = bh / NH, h = bh % NH;
#pragma unroll
    for (int f = 0; f < 4; ++f)
#pragma unroll
        for (int r = 0; r < 8; ++r) {
            const int token = m0 + r + (hi << 3);
            const float v = o[f][r] / lrow[r];
            ao[((size_t)b * Ntok + token) * Cdim + h * HD + f * 16 + col] = f32_bf16(v);
        }
}

// ============================================================================
// Kernel 4: output projection (8192 x 768) @ proj_w^T + proj_b -> f32 out
// ============================================================================
__global__ __launch_bounds__(128) void proj_kernel(
    const __bf16* __restrict__ ao, const float* __restrict__ pw,
    const float* __restrict__ pb, float* __restrict__ out)
{
    __shared__ __align__(16) __bf16 As[64][32];
    __shared__ __align__(16) __bf16 Bs[32][32];
    const int tid = threadIdx.x, wave = tid >> 5, lane = tid & 31;
    const int hi = lane >> 4, col = lane & 15;
    const int m0 = blockIdx.x * 64, n0 = blockIdx.y * 32;
    v8f acc0 = {}, acc1 = {};

    for (int k0 = 0; k0 < Cdim; k0 += 32) {
#pragma unroll
        for (int i = 0; i < 2; ++i) {          // A is already bf16: raw 16B copies
            const int j = tid * 2 + i;         // 256 uint4
            const int r = j >> 2, c = (j & 3) << 3;
            *(uint4*)&As[r][c] = *(const uint4*)&ao[(size_t)(m0 + r) * Cdim + k0 + c];
        }
#pragma unroll
        for (int i = 0; i < 2; ++i) {
            const int j = tid * 2 + i;
            const int n = j >> 3, kq = (j & 7) << 2;
            const float4 f = *(const float4*)&pw[(size_t)(n0 + n) * Cdim + k0 + kq];
            Bs[kq][n]     = f32_bf16(f.x);
            Bs[kq + 1][n] = f32_bf16(f.y);
            Bs[kq + 2][n] = f32_bf16(f.z);
            Bs[kq + 3][n] = f32_bf16(f.w);
        }
        __syncthreads();
        v16bf a  = frag_a_load(&As[wave * 16][0], 32);
        v16bf b0 = frag_b_kn(&Bs[0][0], 32);
        v16bf b1 = frag_b_kn(&Bs[0][16], 32);
        acc0 = wmma_bf16(a, b0, acc0);
        acc1 = wmma_bf16(a, b1, acc1);
        __syncthreads();
    }
#pragma unroll
    for (int s = 0; s < 2; ++s) {
        const v8f acc = s ? acc1 : acc0;
        const int n = n0 + s * 16 + col;
        const float bias = pb[n];
#pragma unroll
        for (int r = 0; r < 8; ++r) {
            const int m = m0 + wave * 16 + r + (hi << 3);
            out[(size_t)m * Cdim + n] = acc[r] + bias;
        }
    }
}

// ============================================================================
extern "C" void kernel_launch(void* const* d_in, const int* in_sizes, int n_in,
                              void* d_out, int out_size, void* d_ws, size_t ws_size,
                              hipStream_t stream) {
    const float* x         = (const float*)d_in[0];
    const float* qkv_w     = (const float*)d_in[1];
    const float* qkv_b     = (const float*)d_in[2];
    const float* proj_w    = (const float*)d_in[3];
    const float* proj_b    = (const float*)d_in[4];
    const float* rel_pos_h = (const float*)d_in[5];
    const float* rel_pos_w = (const float*)d_in[6];
    (void)in_sizes; (void)n_in; (void)out_size; (void)ws_size;

    const size_t QKV_ELEMS = (size_t)BH * Ntok * HD;   // 6,291,456
    const size_t REL_ELEMS = (size_t)BH * 32 * 32 * 32; // 3,145,728
    char* ws = (char*)d_ws;
    __bf16* qb   = (__bf16*)ws;
    __bf16* kb   = qb + QKV_ELEMS;
    __bf16* vb   = kb + QKV_ELEMS;
    float*  relh = (float*)(vb + QKV_ELEMS);
    float*  relw = relh + REL_ELEMS;
    __bf16* ao   = (__bf16*)(relw + REL_ELEMS);

    qkv_kernel<<<dim3(8192 / 64, 2304 / 32), 128, 0, stream>>>(
        x, qkv_w, qkv_b, qb, kb, vb);
    rel_kernel<<<dim3(2 * BH * 32), 32, 0, stream>>>(
        qb, rel_pos_h, rel_pos_w, relh, relw);
    attn_kernel<<<dim3(BH * (Ntok / 64)), 128, 0, stream>>>(
        qb, kb, vb, relh, relw, ao);
    proj_kernel<<<dim3(8192 / 64, Cdim / 32), 128, 0, stream>>>(
        ao, proj_w, proj_b, (float*)d_out);
}
